// TemporalCfCEncoder_21741124452358
// MI455X (gfx1250) — matrix operand order repaired and checked by reading, not compiled, not run on hardware
//
#include <hip/hip_runtime.h>
#include <math.h>

#define DD 1024      // d_model
#define TT 2048      // sequence length
#define FD 4096      // 4*D
#define KW 2048      // 2*D (rows of fused W)
#define NLAYER 4
#define NB 64        // persistent scan blocks
#define SCAN_THREADS 256
#define WSTRIDE 1032 // padded LDS row stride (elements) for W slice

typedef __attribute__((ext_vector_type(16))) __bf16 v16bf;
typedef __attribute__((ext_vector_type(8)))  float  v8f;

union Frag {
    uint4 q[2];
    v16bf v;
};

__device__ __forceinline__ unsigned short f2bf(float f) {
    unsigned u = __float_as_uint(f);
    u += 0x7fffu + ((u >> 16) & 1u);          // round-to-nearest-even
    return (unsigned short)(u >> 16);
}
__device__ __forceinline__ float bflo(unsigned u) { return __uint_as_float(u << 16); }
__device__ __forceinline__ float bfhi(unsigned u) { return __uint_as_float(u & 0xffff0000u); }

// seq0[t][i] = spatial[t][i] + re_norm * re_w[i] + re_b[i]  (stored bf16 as GEMM A)
__global__ void build_seq0_kernel(const float* __restrict__ sp, const float* __restrict__ rn,
                                  const float* __restrict__ rw, const float* __restrict__ rb,
                                  unsigned short* __restrict__ seqbf) {
    int tid = blockIdx.x * 256 + threadIdx.x;     // < TT*DD
    int ii  = tid & (DD - 1);
    float v = sp[tid] + rn[0] * rw[ii] + rb[ii];
    seqbf[tid] = f2bf(v);
}

// Wt[j][k] = concat(Wf1,Wf2,Wta,Wtb)[k][j]  transposed, bf16.  j in [0,4096), k in [0,2048)
__global__ void pack_w_kernel(const float* __restrict__ Wf1, const float* __restrict__ Wf2,
                              const float* __restrict__ Wta, const float* __restrict__ Wtb,
                              unsigned short* __restrict__ Wt) {
    size_t tid = (size_t)blockIdx.x * 256 + threadIdx.x;   // < FD*KW
    int j = (int)(tid % FD);
    int k = (int)(tid / FD);
    int which = j >> 10, jj = j & 1023;
    const float* src = (which == 0) ? Wf1 : (which == 1) ? Wf2 : (which == 2) ? Wta : Wtb;
    Wt[(size_t)j * KW + k] = f2bf(src[(size_t)k * DD + jj]);
}

// Zx[2048][4096] = A[2048][1024](bf16) @ W_x + bias, via v_wmma_f32_16x16x32_bf16.
// 2x2 register blocking per wave: each wave owns a 32x32 output block (4 accumulators),
// reusing each A/B fragment twice -> 2 b128 loads per WMMA instead of 4.
// Block = 8 waves (2x4) -> 64 x 128 block tile. Grid = 32 x 32.
__global__ __launch_bounds__(256) void gemm_zx_kernel(
        const unsigned short* __restrict__ A,   // [2048][1024] bf16 row-major
        const unsigned short* __restrict__ Bt,  // [4096][2048] bf16 (j-major); use k in [0,1024)
        float* __restrict__ Zx,
        const float* __restrict__ b1, const float* __restrict__ b2,
        const float* __restrict__ b3, const float* __restrict__ b4) {
    const int lane = threadIdx.x & 31;
    const int wave = threadIdx.x >> 5;                     // 0..7
    const int row0 = blockIdx.y * 64 + (wave >> 2) * 32;   // output rows (time)
    const int col0 = blockIdx.x * 128 + (wave & 3) * 32;   // output cols (4D)
    const int half = lane >> 4;                            // 0/1
    const int mn   = lane & 15;

    const unsigned short* arow0 = A  + (size_t)(row0 + mn) * DD;
    const unsigned short* arow1 = arow0 + (size_t)16 * DD;
    const unsigned short* brow0 = Bt + (size_t)(col0 + mn) * KW;
    const unsigned short* brow1 = brow0 + (size_t)16 * KW;

    v8f c00 = {}, c01 = {}, c10 = {}, c11 = {};
    #pragma unroll 2
    for (int k0 = 0; k0 < DD; k0 += 32) {
        Frag a0, a1, b0, b1;
        // A 16x32 bf16 layout: VGPR v<4: K = 8*half + 2v,(+1); v>=4: K = 16 + 8*half + 2(v-4),(+1)
        a0.q[0] = *(const uint4*)(arow0 + k0 + 8 * half);
        a0.q[1] = *(const uint4*)(arow0 + k0 + 8 * half + 16);
        a1.q[0] = *(const uint4*)(arow1 + k0 + 8 * half);
        a1.q[1] = *(const uint4*)(arow1 + k0 + 8 * half + 16);
        // B 32x16 bf16 layout: VGPR v<4: K = 8*half + 2v,(+1); v>=4: K = 16 + 8*half + 2(v-4),(+1); N = lane%16 (Bt j-major -> contiguous K)
        b0.q[0] = *(const uint4*)(brow0 + k0 + 8 * half);
        b0.q[1] = *(const uint4*)(brow0 + k0 + 8 * half + 16);
        b1.q[0] = *(const uint4*)(brow1 + k0 + 8 * half);
        b1.q[1] = *(const uint4*)(brow1 + k0 + 8 * half + 16);
        c00 = __builtin_amdgcn_wmma_f32_16x16x32_bf16(false, a0.v, false, b0.v, (short)0, c00, false, false);
        c01 = __builtin_amdgcn_wmma_f32_16x16x32_bf16(false, a0.v, false, b1.v, (short)0, c01, false, false);
        c10 = __builtin_amdgcn_wmma_f32_16x16x32_bf16(false, a1.v, false, b0.v, (short)0, c10, false, false);
        c11 = __builtin_amdgcn_wmma_f32_16x16x32_bf16(false, a1.v, false, b1.v, (short)0, c11, false, false);
    }

    const int colA = col0 + mn;
    const int colB = col0 + 16 + mn;
    const float biasA = (colA < DD) ? b1[colA] : (colA < 2 * DD) ? b2[colA - DD]
                      : (colA < 3 * DD) ? b3[colA - 2 * DD] : b4[colA - 3 * DD];
    const float biasB = (colB < DD) ? b1[colB] : (colB < 2 * DD) ? b2[colB - DD]
                      : (colB < 3 * DD) ? b3[colB - 2 * DD] : b4[colB - 3 * DD];

    // C/D layout: VGPR r -> row (r + 8*half), lane%16 -> col
    float* outA0 = Zx + (size_t)(row0 + 8 * half) * FD + colA;
    float* outB0 = Zx + (size_t)(row0 + 8 * half) * FD + colB;
    float* outA1 = Zx + (size_t)(row0 + 16 + 8 * half) * FD + colA;
    float* outB1 = Zx + (size_t)(row0 + 16 + 8 * half) * FD + colB;
    #pragma unroll
    for (int r = 0; r < 8; ++r) {
        outA0[(size_t)r * FD] = c00[r] + biasA;
        outB0[(size_t)r * FD] = c01[r] + biasB;
        outA1[(size_t)r * FD] = c10[r] + biasA;
        outB1[(size_t)r * FD] = c11[r] + biasB;
    }
}

// Persistent sequential scan. Each block owns hidden indices i in [16b, 16b+16) and the
// corresponding z-columns {i, 1024+i, 2048+i, 3072+i}. W_h slice (64 cols x 1024 k, bf16)
// lives in LDS for the whole kernel. Steps separated by an atomic grid barrier; the
// Zx row for step t+1 is prefetched before the barrier spin to hide its latency.
__global__ __launch_bounds__(SCAN_THREADS) void scan_layer_kernel(
        const float* __restrict__ Zx,           // [2048][4096] fp32 (bias folded in)
        const unsigned short* __restrict__ Wt,  // [4096][2048] bf16; W_h is k in [1024,2048)
        float* hbuf,                            // [1024] fp32, zero-initialized
        unsigned short* __restrict__ seqbf,     // [2048][1024] bf16 out (next layer's A)
        unsigned* cnt,                          // grid barrier counter, zero-initialized
        const float* __restrict__ dtp) {
    __shared__ unsigned short w_lds[64 * WSTRIDE];   // ~129 KB of the 320 KB WGP LDS
    __shared__ float h_lds[DD];
    __shared__ float part[SCAN_THREADS];
    __shared__ float zsh[64];

    const int tid = threadIdx.x;
    const int col_local = tid & 63;          // 0..63 : which*16 + i_local
    const int kc  = tid >> 6;                // 0..3 : k-chunk of 256
    const int which = col_local >> 4;
    const int il    = col_local & 15;
    const int j = (which << 10) + ((int)blockIdx.x << 4) + il;   // global z column
    const float dtv = dtp[0];
    volatile float* hv = hbuf;

    // Stage this block's W_h columns into LDS (once).
    {
        const uint4* src = (const uint4*)(Wt + (size_t)j * KW + DD + kc * 256);
        uint4* dst = (uint4*)(&w_lds[col_local * WSTRIDE + kc * 256]);
        #pragma unroll
        for (int it = 0; it < 32; ++it) dst[it] = src[it];
    }
    __syncthreads();

    const uint4* wrow = (const uint4*)(&w_lds[col_local * WSTRIDE + kc * 256]);
    const int kb = kc << 8;

    float zx_pref = (tid < 64) ? Zx[j] : 0.f;   // Zx row for t=0

    for (int t = 0; t < TT; ++t) {
        // broadcast current h into LDS
        for (int idx = tid; idx < DD; idx += SCAN_THREADS) h_lds[idx] = hv[idx];
        __syncthreads();

        float acc = 0.f;
        #pragma unroll 8
        for (int it = 0; it < 32; ++it) {
            uint4 w = wrow[it];
            const float* hp = &h_lds[kb + (it << 3)];
            acc += bflo(w.x) * hp[0] + bfhi(w.x) * hp[1]
                 + bflo(w.y) * hp[2] + bfhi(w.y) * hp[3]
                 + bflo(w.z) * hp[4] + bfhi(w.z) * hp[5]
                 + bflo(w.w) * hp[6] + bfhi(w.w) * hp[7];
        }
        part[tid] = acc;
        __syncthreads();

        if (tid < 64) {   // kc==0 lanes: finish z = Zx[t][j] + h.W_h[:,j]
            zsh[tid] = zx_pref + part[tid] + part[64 + tid] + part[128 + tid] + part[192 + tid];
        }
        __syncthreads();

        if (tid < 16) {
            float zf1 = zsh[tid],      zf2 = zsh[16 + tid];
            float zta = zsh[32 + tid], ztb = zsh[48 + tid];
            float g  = 1.f / (1.f + expf(zta * dtv - ztb));   // sigmoid(-zta*dt + ztb)
            float hn = g * tanhf(zf1) + (1.f - g) * tanhf(zf2);
            int i = ((int)blockIdx.x << 4) + tid;
            seqbf[(size_t)t * DD + i] = f2bf(hn);
            hv[i] = hn;
        }
        __threadfence();

        // prefetch next step's Zx element; latency hides behind the barrier spin
        float zx_next = 0.f;
        if (tid < 64 && (t + 1) < TT) zx_next = Zx[(size_t)(t + 1) * FD + j];
        __syncthreads();

        // grid barrier (monotonic counter; reset by memset before launch)
        if (tid == 0) {
            __hip_atomic_fetch_add(cnt, 1u, __ATOMIC_ACQ_REL, __HIP_MEMORY_SCOPE_AGENT);
            const unsigned target = (unsigned)(t + 1) * gridDim.x;
            while (__hip_atomic_load(cnt, __ATOMIC_ACQUIRE, __HIP_MEMORY_SCOPE_AGENT) < target) {
                __builtin_amdgcn_s_sleep(1);
            }
        }
        __syncthreads();
        zx_pref = zx_next;
    }
}

__global__ void copy_out_kernel(const float* __restrict__ h, float* __restrict__ out) {
    int i = blockIdx.x * 256 + threadIdx.x;
    if (i < DD) out[i] = h[i];
}

extern "C" void kernel_launch(void* const* d_in, const int* in_sizes, int n_in,
                              void* d_out, int out_size, void* d_ws, size_t ws_size,
                              hipStream_t stream) {
    (void)in_sizes; (void)n_in; (void)out_size;

    const float* sp  = (const float*)d_in[0];   // spatial_states [T,D]
    const float* rn  = (const float*)d_in[1];   // re_norm scalar
    const float* dt  = (const float*)d_in[2];   // dt_phys scalar
    const float* rw  = (const float*)d_in[3];   // re_proj_w [D]
    const float* rb  = (const float*)d_in[4];   // re_proj_b [D]
    const float* Wf1 = (const float*)d_in[5];
    const float* bf1 = (const float*)d_in[6];
    const float* Wf2 = (const float*)d_in[7];
    const float* bf2 = (const float*)d_in[8];
    const float* Wta = (const float*)d_in[9];
    const float* bta = (const float*)d_in[10];
    const float* Wtb = (const float*)d_in[11];
    const float* btb = (const float*)d_in[12];

    // workspace layout
    const size_t o_Wt  = 0;                     // 4096*2048*2  = 16 MB
    const size_t o_Zx  = (size_t)16 << 20;      // 2048*4096*4  = 32 MB
    const size_t o_seq = (size_t)48 << 20;      // 2048*1024*2  =  4 MB
    const size_t o_h   = (size_t)52 << 20;      // 4 KB
    const size_t o_cnt = o_h + 4096;            // 4 B
    if (ws_size < o_cnt + 64) return;

    char* ws = (char*)d_ws;
    unsigned short* Wt    = (unsigned short*)(ws + o_Wt);
    float*          Zx    = (float*)(ws + o_Zx);
    unsigned short* seqbf = (unsigned short*)(ws + o_seq);
    float*          hbuf  = (float*)(ws + o_h);
    unsigned*       cnt   = (unsigned*)(ws + o_cnt);

    build_seq0_kernel<<<(TT * DD) / 256, 256, 0, stream>>>(sp, rn, rw, rb, seqbf);

    for (int l = 0; l < NLAYER; ++l) {
        const size_t wo = (size_t)l * KW * DD;   // per-layer weight offset (elements)
        pack_w_kernel<<<(FD * KW) / 256, 256, 0, stream>>>(Wf1 + wo, Wf2 + wo, Wta + wo, Wtb + wo, Wt);
        gemm_zx_kernel<<<dim3(FD / 128, TT / 64), 256, 0, stream>>>(
            seqbf, Wt, Zx, bf1 + l * DD, bf2 + l * DD, bta + l * DD, btb + l * DD);
        hipMemsetAsync(hbuf, 0, DD * sizeof(float), stream);
        hipMemsetAsync(cnt, 0, sizeof(unsigned), stream);
        scan_layer_kernel<<<NB, SCAN_THREADS, 0, stream>>>(Zx, Wt, hbuf, seqbf, cnt, dt);
    }

    copy_out_kernel<<<(DD + 255) / 256, 256, 0, stream>>>(hbuf, (float*)d_out);
}
